// RTTransformer_5652176962161
// MI455X (gfx1250) — matrix operand order repaired and checked by reading, not compiled, not run on hardware
//
#include <hip/hip_runtime.h>
#include <math.h>

typedef __attribute__((ext_vector_type(16))) _Float16 v16h;
typedef __attribute__((ext_vector_type(8)))  _Float16 v8h;
typedef __attribute__((ext_vector_type(8)))  float    v8f;
typedef __attribute__((ext_vector_type(8)))  float    v8fv;

typedef union { v16h v; v8h h[2]; } AB16;

#define NNODE 67
#define BATCH 32

__device__ inline v8f v8f_zero() { v8f z;
#pragma unroll
  for (int q = 0; q < 8; ++q) z[q] = 0.f; return z; }

__device__ inline float gelu_exact(float x) {
  return 0.5f * x * (1.f + erff(x * 0.70710678118654752f));
}

// ---------------------------------------------------------------------------
// Generic WMMA GEMM: C[M,Nout] = A[M,KT](f16) @ WT[Nout,KT](f16,transposed)
// modes: 0: f16 out = acc+bias
//        1: f16 out = gelu(acc+bias)
//        2: f32 out = acc+bias
//        3: f32 out += acc+bias   (residual accumulate)
//        4: f16 out = gelu(acc+bias+S[row/N]+T[b*N + row%N])   (edge mix)
// A1 (optional) supplies the K>=128 half of a concat A (both lda=128).
// Out-of-range A rows are CLAMPED (not zeroed): a garbage A row only affects
// the same out-of-range D row, which is masked at the store.
// ---------------------------------------------------------------------------
template <int KT>
__global__ __launch_bounds__(128) void gemm_wmma(
    const _Float16* __restrict__ A0, int lda0,
    const _Float16* __restrict__ A1,
    const _Float16* __restrict__ WT, int Nout, int M,
    const float* __restrict__ bias, void* __restrict__ outp, int mode,
    const float* __restrict__ Srow, const float* __restrict__ Tcol, int Nedge)
{
  const int lane = threadIdx.x & 31;
  const int wid  = threadIdx.x >> 5;
  const long m0  = (long)blockIdx.x * 64 + (long)wid * 16;
  const int  c0  = blockIdx.y * 64;

  v8f acc[4];
#pragma unroll
  for (int t = 0; t < 4; ++t) acc[t] = v8f_zero();

  long arow = m0 + (lane & 15);
  if (arow >= (long)M) arow = (long)M - 1;   // clamp; masked at store

#pragma unroll
  for (int kk = 0; kk < (KT >> 5); ++kk) {
    const int kb = kk * 32 + ((lane >> 4) << 4);
    const _Float16* ap; int lda; int kloc;
    if (A1 && kb >= 128) { ap = A1; lda = 128; kloc = kb - 128; }
    else                 { ap = A0; lda = lda0; kloc = kb; }
    AB16 a;
    const _Float16* base = ap + (size_t)arow * lda + kloc;
    a.h[0] = *(const v8h*)(base);
    a.h[1] = *(const v8h*)(base + 8);

    const int kbb = kk * 32 + ((lane >> 4) << 4);
#pragma unroll
    for (int t = 0; t < 4; ++t) {
      int n = c0 + t * 16 + (lane & 15);
      v16h b = *(const v16h*)(WT + (size_t)n * KT + kbb);
      acc[t] = __builtin_amdgcn_wmma_f32_16x16x32_f16(false, a.v, false, b,
                                                      (short)0, acc[t], false, false);
    }
  }

  const int hrow = (lane >> 4) * 8;
  const int coll = lane & 15;
#pragma unroll
  for (int t = 0; t < 4; ++t) {
    int col = c0 + t * 16 + coll;
    float bv = bias ? bias[col] : 0.f;
#pragma unroll
    for (int r = 0; r < 8; ++r) {
      long row = m0 + hrow + r;
      if (row < (long)M) {
        float x = acc[t][r] + bv;
        if (mode == 4) {
          long bi = row / Nedge;                    // b*N + i
          long j  = row % Nedge;
          long bb = row / ((long)Nedge * Nedge);    // b
          x += Srow[bi * 128 + col] + Tcol[(bb * Nedge + j) * 128 + col];
          ((_Float16*)outp)[(size_t)row * Nout + col] = (_Float16)gelu_exact(x);
        } else if (mode == 1) {
          ((_Float16*)outp)[(size_t)row * Nout + col] = (_Float16)gelu_exact(x);
        } else if (mode == 0) {
          ((_Float16*)outp)[(size_t)row * Nout + col] = (_Float16)x;
        } else if (mode == 2) {
          ((float*)outp)[(size_t)row * Nout + col] = x;
        } else { // 3
          ((float*)outp)[(size_t)row * Nout + col] += x;
        }
      }
    }
  }
}

// ---------------------------------------------------------------------------
// Fused attention per (b,i): WMMA computes qkv_e = edge[b,i,:,:] @ Wqkv_e^T
// into LDS, then per-head softmax attention -> o16[b,i,:]
// ---------------------------------------------------------------------------
__global__ __launch_bounds__(480) void attn_kernel(
    const float* __restrict__ edge, const float* __restrict__ qkvn,
    const _Float16* __restrict__ WqeT, _Float16* __restrict__ o16)
{
  __shared__ _Float16 qkv[68 * 384];
  __shared__ float attw[8][80];

  const int blk = blockIdx.x;
  const int b = blk / NNODE, i = blk % NNODE;
  const int lane = threadIdx.x & 31;
  const int wid  = threadIdx.x >> 5;
  const float* erow = edge + (size_t)(b * NNODE + i) * NNODE * 128;

  { // GEMM phase: 15 waves = 5 m-tiles x 3 col-blocks(128)
    const int mt = wid / 3, cb = wid % 3;
    const int m0 = mt * 16, c0 = cb * 128;
    v8f acc[8];
#pragma unroll
    for (int t = 0; t < 8; ++t) acc[t] = v8f_zero();
    int j = m0 + (lane & 15);
    if (j >= NNODE) j = NNODE - 1;                 // clamp; masked at LDS store
#pragma unroll
    for (int kk = 0; kk < 4; ++kk) {
      const int kb = kk * 32 + ((lane >> 4) << 4);
      const float* base = erow + (size_t)j * 128 + kb;
      v8fv f0 = *(const v8fv*)(base);
      v8fv f1 = *(const v8fv*)(base + 8);
      AB16 a;
      a.h[0] = __builtin_convertvector(f0, v8h);
      a.h[1] = __builtin_convertvector(f1, v8h);
      const int kbb = kk * 32 + ((lane >> 4) << 4);
#pragma unroll
      for (int t = 0; t < 8; ++t) {
        int n = c0 + t * 16 + (lane & 15);
        v16h bf = *(const v16h*)(WqeT + (size_t)n * 128 + kbb);
        acc[t] = __builtin_amdgcn_wmma_f32_16x16x32_f16(false, a.v, false, bf,
                                                        (short)0, acc[t], false, false);
      }
    }
    const int hrow = (lane >> 4) * 8;
    const int coll = lane & 15;
#pragma unroll
    for (int t = 0; t < 8; ++t) {
      int col = c0 + t * 16 + coll;
#pragma unroll
      for (int r = 0; r < 8; ++r) {
        int j2 = m0 + hrow + r;
        if (j2 < NNODE) qkv[j2 * 384 + col] = (_Float16)acc[t][r];
      }
    }
  }
  __syncthreads();

  if (wid < 8) { // score phase: one wave per head
    const int h = wid;
    const float* qn = qkvn + (size_t)(b * NNODE + i) * 384 + h * 48;
    float sv[3];
    float mx = -1e30f;
#pragma unroll
    for (int t = 0; t < 3; ++t) {
      int j = lane + t * 32;
      float s = -1e30f;
      if (j < NNODE) {
        const float* kn = qkvn + (size_t)(b * NNODE + j) * 384 + h * 48 + 16;
        float d = 0.f;
#pragma unroll
        for (int dh = 0; dh < 16; ++dh) {
          float q = qn[dh] + (float)qkv[j * 384 + h * 48 + dh];
          float k = kn[dh] + (float)qkv[j * 384 + h * 48 + 16 + dh];
          d += q * k;
        }
        s = d * 0.25f;   // 1/sqrt(16)
      }
      sv[t] = s;
      mx = fmaxf(mx, s);
    }
    for (int m = 16; m >= 1; m >>= 1) mx = fmaxf(mx, __shfl_xor(mx, m, 32));
    float sum = 0.f;
#pragma unroll
    for (int t = 0; t < 3; ++t) {
      int j = lane + t * 32;
      float e = (j < NNODE) ? __expf(sv[t] - mx) : 0.f;
      sv[t] = e; sum += e;
    }
    for (int m = 16; m >= 1; m >>= 1) sum += __shfl_xor(sum, m, 32);
    float inv = 1.f / sum;
#pragma unroll
    for (int t = 0; t < 3; ++t) { int j = lane + t * 32; if (j < NNODE) attw[h][j] = sv[t] * inv; }
  }
  __syncthreads();
  if (wid < 8) { // output phase
    const int h = wid;
    const int dh = lane & 15, jp = lane >> 4;
    float acc = 0.f;
    for (int j = jp; j < NNODE; j += 2) {
      float v = qkvn[(size_t)(b * NNODE + j) * 384 + h * 48 + 32 + dh]
              + (float)qkv[j * 384 + h * 48 + 32 + dh];
      acc += attw[h][j] * v;
    }
    acc += __shfl_xor(acc, 16, 32);
    if (lane < 16) o16[(size_t)(b * NNODE + i) * 128 + h * 16 + dh] = (_Float16)acc;
  }
}

// ---------------------------------------------------------------------------
// LayerNorm over D=128, one wave per row, f16 output
// ---------------------------------------------------------------------------
__global__ void ln_kernel(const float* __restrict__ x, const float* __restrict__ g,
                          const float* __restrict__ bt, _Float16* __restrict__ out, long rows)
{
  const int lane = threadIdx.x & 31, wid = threadIdx.x >> 5;
  const long row = (long)blockIdx.x * 8 + wid;
  if (row >= rows) return;
  const float* xr = x + row * 128;
  float v0 = xr[lane], v1 = xr[lane + 32], v2 = xr[lane + 64], v3 = xr[lane + 96];
  float s  = v0 + v1 + v2 + v3;
  float s2 = v0 * v0 + v1 * v1 + v2 * v2 + v3 * v3;
  for (int m = 16; m >= 1; m >>= 1) { s += __shfl_xor(s, m, 32); s2 += __shfl_xor(s2, m, 32); }
  float mean = s * (1.f / 128.f);
  float var  = s2 * (1.f / 128.f) - mean * mean;
  float inv  = rsqrtf(var + 1e-5f);
  _Float16* o = out + row * 128;
  o[lane]      = (_Float16)((v0 - mean) * inv * g[lane]      + bt[lane]);
  o[lane + 32] = (_Float16)((v1 - mean) * inv * g[lane + 32] + bt[lane + 32]);
  o[lane + 64] = (_Float16)((v2 - mean) * inv * g[lane + 64] + bt[lane + 64]);
  o[lane + 96] = (_Float16)((v3 - mean) * inv * g[lane + 96] + bt[lane + 96]);
}

__global__ void wtk(const float* __restrict__ src, _Float16* __restrict__ dst, int K, int Nn)
{
  int idx = blockIdx.x * 256 + threadIdx.x;
  if (idx >= K * Nn) return;
  int n = idx / K, k = idx % K;
  dst[idx] = (_Float16)src[(size_t)k * Nn + n];
}

__global__ void cvt16(const float* __restrict__ x, _Float16* __restrict__ y, long n)
{
  long i = (long)blockIdx.x * 256 + threadIdx.x;
  if (i < n) y[i] = (_Float16)x[i];
}

__global__ void precomp(const float* pw_W, const float* pw_b, const float* pb_W, const float* pb_b,
                        const float* niW, const float* nib, const float* eiW, const float* eib,
                        float* p1, float* p0, float* q1, float* q0)
{
  int d = threadIdx.x;
  float a = 0, bb = 0, c = 0, e = 0;
  for (int k = 0; k < 128; ++k) {
    a  += pw_W[k] * eiW[k * 128 + d];
    bb += pw_b[k] * eiW[k * 128 + d];
    c  += pb_W[k] * niW[k * 128 + d];
    e  += pb_b[k] * niW[k * 128 + d];
  }
  p1[d] = a; p0[d] = bb + eib[d]; q1[d] = c; q0[d] = e + nib[d];
}

__global__ void edge_fill(const float* __restrict__ w0, const float* __restrict__ w1,
                          const float* __restrict__ w2, const float* __restrict__ p1,
                          const float* __restrict__ p0, float* __restrict__ edge)
{
  long r = blockIdx.x; int d = threadIdx.x;
  int b = (int)(r / (NNODE * NNODE)); int rem = (int)(r % (NNODE * NNODE));
  int i = rem / NNODE, j = rem % NNODE;
  float u = 0.f;
  if (i < 2 && j >= 2 && j < 34)                 u = w0[(b * 2 + i) * 32 + (j - 2)];
  else if (i >= 2 && i < 34 && j >= 34 && j < 66) u = w1[(b * 32 + (i - 2)) * 32 + (j - 34)];
  else if (i >= 34 && i < 66 && j == 66)          u = w2[b * 32 + (i - 34)];
  edge[r * 128 + d] = u * p1[d] + p0[d];
}

__global__ void node_fill(const float* __restrict__ b0, const float* __restrict__ b1,
                          const float* __restrict__ b2, const float* __restrict__ q1,
                          const float* __restrict__ q0, const float* __restrict__ pos,
                          float* __restrict__ node)
{
  int blk = blockIdx.x; int d = threadIdx.x;
  int b = blk / NNODE, i = blk % NNODE;
  float v = 0.f;
  if (i >= 2 && i < 34)       v = b0[b * 32 + (i - 2)];
  else if (i >= 34 && i < 66) v = b1[b * 32 + (i - 34)];
  else if (i == 66)           v = b2[b];
  int seg = (i == 0) ? 0 : (i == 1) ? 1 : (i < 34) ? 2 : (i < 66) ? 3 : 4;
  node[(size_t)blk * 128 + d] = v * q1[d] + q0[d] + pos[seg * 128 + d];
}

__global__ void rowmean16(const _Float16* __restrict__ en, _Float16* __restrict__ out)
{
  int bi = blockIdx.x; int d = threadIdx.x;
  const _Float16* base = en + (size_t)bi * NNODE * 128 + d;
  float s = 0.f;
  for (int j = 0; j < NNODE; ++j) s += (float)base[(size_t)j * 128];
  out[(size_t)bi * 128 + d] = (_Float16)(s * (1.f / NNODE));
}

__global__ void colmean16(const _Float16* __restrict__ en, _Float16* __restrict__ out)
{
  int blk = blockIdx.x; int b = blk / NNODE, j = blk % NNODE; int d = threadIdx.x;
  const _Float16* base = en + ((size_t)b * NNODE * NNODE + j) * 128 + d;
  float s = 0.f;
  for (int i = 0; i < NNODE; ++i) s += (float)base[(size_t)i * NNODE * 128];
  out[(size_t)blk * 128 + d] = (_Float16)(s * (1.f / NNODE));
}

__global__ void etranspose(const _Float16* __restrict__ en, _Float16* __restrict__ enT)
{
  long r = blockIdx.x; int d = threadIdx.x;
  int b = (int)(r / (NNODE * NNODE)); int rem = (int)(r % (NNODE * NNODE));
  int i = rem / NNODE, j = rem % NNODE;
  enT[r * 128 + d] = en[(((size_t)b * NNODE + j) * NNODE + i) * 128 + d];
}

__global__ void head_kernel(const float* __restrict__ node, const float* __restrict__ Wo1,
                            const float* __restrict__ bo1, const float* __restrict__ Wo2,
                            const float* __restrict__ bo2, float* __restrict__ out)
{
  __shared__ float g[128]; __shared__ float h1[128];
  int b = blockIdx.x; int d = threadIdx.x;
  float s = 0.f;
  for (int i = 0; i < NNODE; ++i) s += node[((size_t)b * NNODE + i) * 128 + d];
  g[d] = s * (1.f / NNODE);
  __syncthreads();
  float a = bo1[d];
  for (int k = 0; k < 128; ++k) a += g[k] * Wo1[k * 128 + d];
  h1[d] = fmaxf(a, 0.f);
  __syncthreads();
  if (d < 10) {
    float a2 = bo2[d];
    for (int k = 0; k < 128; ++k) a2 += h1[k] * Wo2[k * 10 + d];
    out[b * 10 + d] = a2;
  }
}

// ---------------------------------------------------------------------------

extern "C" void kernel_launch(void* const* d_in, const int* in_sizes, int n_in,
                              void* d_out, int out_size, void* d_ws, size_t ws_size,
                              hipStream_t stream)
{
  (void)in_sizes; (void)n_in; (void)out_size; (void)ws_size;
  const float* w0     = (const float*)d_in[0];
  const float* w1     = (const float*)d_in[1];
  const float* w2     = (const float*)d_in[2];
  const float* b0     = (const float*)d_in[3];
  const float* b1     = (const float*)d_in[4];
  const float* b2     = (const float*)d_in[5];
  const float* pw_W   = (const float*)d_in[6];
  const float* pw_b   = (const float*)d_in[7];
  const float* pb_W   = (const float*)d_in[8];
  const float* pb_b   = (const float*)d_in[9];
  const float* niW    = (const float*)d_in[10];
  const float* nib    = (const float*)d_in[11];
  const float* eiW    = (const float*)d_in[12];
  const float* eib    = (const float*)d_in[13];
  const float* pos    = (const float*)d_in[14];
  const float* aln_g  = (const float*)d_in[15];
  const float* aln_b  = (const float*)d_in[16];
  const float* Wqkv_n = (const float*)d_in[17];
  const float* Wqkv_e = (const float*)d_in[18];
  const float* Wao    = (const float*)d_in[19];
  const float* bao    = (const float*)d_in[20];
  const float* Wl0    = (const float*)d_in[21];
  const float* bl0    = (const float*)d_in[22];
  const float* nln_g  = (const float*)d_in[23];
  const float* nln_b  = (const float*)d_in[24];
  const float* nW1    = (const float*)d_in[25];
  const float* nb1    = (const float*)d_in[26];
  const float* nW2    = (const float*)d_in[27];
  const float* nb2    = (const float*)d_in[28];
  const float* eln_g  = (const float*)d_in[29];
  const float* eln_b  = (const float*)d_in[30];
  const float* We     = (const float*)d_in[31];
  const float* be     = (const float*)d_in[32];
  const float* Ws     = (const float*)d_in[33];
  const float* bs     = (const float*)d_in[34];
  const float* Wt     = (const float*)d_in[35];
  const float* bt     = (const float*)d_in[36];
  const float* Wer    = (const float*)d_in[37];
  const float* Wec    = (const float*)d_in[38];
  const float* We1    = (const float*)d_in[39];
  const float* be1    = (const float*)d_in[40];
  const float* Wel0   = (const float*)d_in[41];
  const float* bel0   = (const float*)d_in[42];
  const float* mln_g  = (const float*)d_in[43];
  const float* mln_b  = (const float*)d_in[44];
  const float* mW1    = (const float*)d_in[45];
  const float* mb1    = (const float*)d_in[46];
  const float* mW2    = (const float*)d_in[47];
  const float* mb2    = (const float*)d_in[48];
  const float* Wo1    = (const float*)d_in[49];
  const float* bo1    = (const float*)d_in[50];
  const float* Wo2    = (const float*)d_in[51];
  const float* bo2    = (const float*)d_in[52];

  const long BN  = (long)BATCH * NNODE;            // 2144
  const long BNN = (long)BATCH * NNODE * NNODE;    // 143648

  char* p = (char*)d_ws;
  auto alloc = [&](size_t bytes) -> char* {
    char* r = p; p += (bytes + 255) & ~(size_t)255; return r;
  };

  float*    edge  = (float*)alloc(BNN * 128 * 4);
  _Float16* F0    = (_Float16*)alloc(BNN * 256 * 2);   // up to 256-wide f16 edge buf
  _Float16* F1    = (_Float16*)alloc(BNN * 128 * 2);
  _Float16* F2    = (_Float16*)alloc(BNN * 128 * 2);
  float*    node  = (float*)alloc(BN * 128 * 4);
  float*    qkvn  = (float*)alloc(BN * 384 * 4);
  _Float16* h16   = (_Float16*)alloc(BN * 128 * 2);
  _Float16* o16   = (_Float16*)alloc(BN * 128 * 2);
  _Float16* tmp16 = (_Float16*)alloc(BN * 128 * 2);
  _Float16* u16n  = (_Float16*)alloc(BN * 256 * 2);
  _Float16* nd16  = (_Float16*)alloc(BN * 128 * 2);
  _Float16* enr16 = (_Float16*)alloc(BN * 128 * 2);
  _Float16* enc16 = (_Float16*)alloc(BN * 128 * 2);
  float*    SP    = (float*)alloc(BN * 128 * 4);
  float*    TP    = (float*)alloc(BN * 128 * 4);
  float*    p1v   = (float*)alloc(128 * 4);
  float*    p0v   = (float*)alloc(128 * 4);
  float*    q1v   = (float*)alloc(128 * 4);
  float*    q0v   = (float*)alloc(128 * 4);

  auto prepW = [&](const float* src, int K, int Nn) -> _Float16* {
    _Float16* dst = (_Float16*)alloc((size_t)K * Nn * 2);
    int total = K * Nn;
    wtk<<<(total + 255) / 256, 256, 0, stream>>>(src, dst, K, Nn);
    return dst;
  };

  _Float16 *WqnT[3], *WqeT[3], *WaoT[3], *Wl0T[3], *nW1T[3], *nW2T[3], *WeT[3],
           *WsT[3], *WtT[3], *WerT[3], *WecT[3], *We1T[3], *Wel0T[3], *mW1T[3], *mW2T[3];
  for (int l = 0; l < 3; ++l) {
    WqnT[l]  = prepW(Wqkv_n + (size_t)l * 49152, 128, 384);
    WqeT[l]  = prepW(Wqkv_e + (size_t)l * 49152, 128, 384);
    WaoT[l]  = prepW(Wao    + (size_t)l * 16384, 128, 128);
    Wl0T[l]  = prepW(Wl0    + (size_t)l * 16384, 128, 128);
    nW1T[l]  = prepW(nW1    + (size_t)l * 32768, 128, 256);
    nW2T[l]  = prepW(nW2    + (size_t)l * 32768, 256, 128);
    WeT[l]   = prepW(We     + (size_t)l * 32768, 256, 128);
    WsT[l]   = prepW(Ws     + (size_t)l * 16384, 128, 128);
    WtT[l]   = prepW(Wt     + (size_t)l * 16384, 128, 128);
    WerT[l]  = prepW(Wer    + (size_t)l * 16384, 128, 128);
    WecT[l]  = prepW(Wec    + (size_t)l * 16384, 128, 128);
    We1T[l]  = prepW(We1    + (size_t)l * 16384, 128, 128);
    Wel0T[l] = prepW(Wel0   + (size_t)l * 16384, 128, 128);
    mW1T[l]  = prepW(mW1    + (size_t)l * 32768, 128, 256);
    mW2T[l]  = prepW(mW2    + (size_t)l * 32768, 256, 128);
  }

  auto gemm = [&](const _Float16* A0, int lda0, const _Float16* A1,
                  const _Float16* WT, int K, int Nout, long M,
                  const float* bias, void* out, int mode,
                  const float* S = nullptr, const float* T = nullptr) {
    dim3 grid((unsigned)((M + 63) / 64), (unsigned)(Nout / 64));
    if (K == 128)
      gemm_wmma<128><<<grid, dim3(128), 0, stream>>>(A0, lda0, A1, WT, Nout, (int)M,
                                                     bias, out, mode, S, T, NNODE);
    else
      gemm_wmma<256><<<grid, dim3(128), 0, stream>>>(A0, lda0, A1, WT, Nout, (int)M,
                                                     bias, out, mode, S, T, NNODE);
  };
  auto ln = [&](const float* x, const float* g, const float* b, _Float16* out, long rows) {
    ln_kernel<<<(unsigned)((rows + 7) / 8), 256, 0, stream>>>(x, g, b, out, rows);
  };

  // ---- embedding (rank-1) ----
  precomp<<<1, 128, 0, stream>>>(pw_W, pw_b, pb_W, pb_b, niW, nib, eiW, eib,
                                 p1v, p0v, q1v, q0v);
  edge_fill<<<(unsigned)BNN, 128, 0, stream>>>(w0, w1, w2, p1v, p0v, edge);
  node_fill<<<(unsigned)BN, 128, 0, stream>>>(b0, b1, b2, q1v, q0v, pos, node);

  // ---- layers ----
  for (int l = 0; l < 3; ++l) {
    // node attention
    ln(node, aln_g + l * 128, aln_b + l * 128, h16, BN);
    gemm(h16, 128, nullptr, WqnT[l], 128, 384, BN, nullptr, qkvn, 2);
    attn_kernel<<<(unsigned)BN, 480, 0, stream>>>(edge, qkvn, WqeT[l], o16);
    gemm(o16, 128, nullptr, WaoT[l], 128, 128, BN, bao + l * 128, tmp16, 0);
    gemm(tmp16, 128, nullptr, Wl0T[l], 128, 128, BN, bl0 + l * 128, node, 3);
    // node MLP
    ln(node, nln_g + l * 128, nln_b + l * 128, h16, BN);
    gemm(h16, 128, nullptr, nW1T[l], 128, 256, BN, nb1 + l * 256, u16n, 1);
    gemm(u16n, 256, nullptr, nW2T[l], 256, 128, BN, nb2 + l * 128, node, 3);

    // edge mixing
    ln(edge, eln_g + l * 128, eln_b + l * 128, F1, BNN);              // en
    etranspose<<<(unsigned)BNN, 128, 0, stream>>>(F1, F2);            // en^T
    rowmean16<<<(unsigned)BN, 128, 0, stream>>>(F1, enr16);
    colmean16<<<(unsigned)BN, 128, 0, stream>>>(F1, enc16);
    cvt16<<<(unsigned)((BN * 128 + 255) / 256), 256, 0, stream>>>(node, nd16, BN * 128);
    gemm(nd16, 128, nullptr, WsT[l], 128, 128, BN, bs + l * 128, SP, 2);
    gemm(enr16, 128, nullptr, WerT[l], 128, 128, BN, nullptr, SP, 3);
    gemm(nd16, 128, nullptr, WtT[l], 128, 128, BN, bt + l * 128, TP, 2);
    gemm(enc16, 128, nullptr, WecT[l], 128, 128, BN, nullptr, TP, 3);
    // gelu(concat(en,enT)@We + be + SP[i] + TP[j]) -> F0 (128-wide)
    gemm(F1, 128, F2, WeT[l], 256, 128, BNN, be + l * 128, F0, 4, SP, TP);
    gemm(F0, 128, nullptr, We1T[l], 128, 128, BNN, be1 + l * 128, F1, 0);
    gemm(F1, 128, nullptr, Wel0T[l], 128, 128, BNN, bel0 + l * 128, edge, 3);
    // edge MLP
    ln(edge, mln_g + l * 128, mln_b + l * 128, F1, BNN);
    gemm(F1, 128, nullptr, mW1T[l], 128, 256, BNN, mb1 + l * 256, F0, 1);
    gemm(F0, 256, nullptr, mW2T[l], 256, 128, BNN, mb2 + l * 128, edge, 3);
  }

  // ---- head ----
  head_kernel<<<BATCH, 128, 0, stream>>>(node, Wo1, bo1, Wo2, bo2, (float*)d_out);
}
